// NMT_85581518340682
// MI455X (gfx1250) — matrix operand order repaired and checked by reading, not compiled, not run on hardware
//
#include <hip/hip_runtime.h>
#include <hip/hip_bf16.h>
#include <math.h>

// Problem dims (from reference)
#define TSTEPS 127      // T-1 recurrent steps
#define BB 32           // batch
#define VV 32000        // vocab
#define EE 512          // embed dim
#define HH 1024         // hidden
#define G4 4096         // 4*H
#define NWG_LOGITS (VV / 128)   // 250 workgroups, 128 vocab columns each

typedef __attribute__((ext_vector_type(16))) __bf16 v16bf;
typedef __attribute__((ext_vector_type(8)))  float  v8f;

union ABu { uint32_t u[8]; v16bf v; };

// ---- WMMA tile loaders (contiguous k = 16*(lane/16) + s in A and B, 16-bit A 16x32) ----

// A tile: 16x32 (MxK) from row-major src with leading dim ld.
__device__ __forceinline__ v16bf load_a_tile(const __bf16* __restrict__ src,
                                             int ld, int m0, int k0, int lane) {
  ABu a;
  const int m     = lane & 15;
  const int kbase = (lane < 16) ? 0 : 16;
  const __bf16* row = src + (size_t)(m0 + m) * ld + k0;
#pragma unroll
  for (int v = 0; v < 8; ++v) {
    const int kk = kbase + 2 * v;
    a.u[v] = *(const uint32_t*)(row + kk);
  }
  return a.v;
}

// B tile: 32x16 (KxN) where B[k][n] = W[n0+n][k0+k], W row-major (ldw elems).
__device__ __forceinline__ v16bf load_b_tile(const __bf16* __restrict__ W,
                                             int ldw, int n0, int k0, int lane) {
  ABu b;
  const int n  = lane & 15;
  const int kb = (lane < 16) ? 0 : 16;
  const __bf16* wrow = W + (size_t)(n0 + n) * ldw + k0 + kb;
  __builtin_prefetch(wrow + 2 * ldw, 0, 0);   // global_prefetch_b8: next rows
#pragma unroll
  for (int v = 0; v < 8; ++v) b.u[v] = *(const uint32_t*)(wrow + 2 * v);
  return b.v;
}

// ---- setup kernels ----

__global__ void k_zero(float* out) { out[0] = 0.0f; }

__global__ void k_cvt_bf16(const float* __restrict__ src, __bf16* __restrict__ dst, int n) {
  int i = blockIdx.x * blockDim.x + threadIdx.x;
  if (i < n) dst[i] = (__bf16)src[i];
}

// x_all[t][b][e] = relu(emb[indices[t][b]][e]) as bf16, t in [0,126]
__global__ void k_embed(const int* __restrict__ idx, const float* __restrict__ emb,
                        __bf16* __restrict__ xall) {
  int i = blockIdx.x * blockDim.x + threadIdx.x;
  const int tot = TSTEPS * BB * EE;
  if (i >= tot) return;
  int e = i & (EE - 1);
  int r = i >> 9;            // EE = 512
  int b = r & (BB - 1);
  int t = r >> 5;            // BB = 32
  int v = idx[t * BB + b];
  float x = emb[(size_t)v * EE + e];
  xall[i] = (__bf16)fmaxf(x, 0.0f);
}

// ---- gates GEMM: gates(32,4096) = x(32,512)@Wih^T + h(32,1024)@Whh^T + b ----
__global__ void __launch_bounds__(256)
k_gates(const __bf16* __restrict__ x, const __bf16* __restrict__ h,
        const __bf16* __restrict__ Wih, const __bf16* __restrict__ Whh,
        const float* __restrict__ b_ih, const float* __restrict__ b_hh,
        float* __restrict__ gates) {
  const int lane = threadIdx.x & 31;
  const int wave = threadIdx.x >> 5;
  const int n0   = blockIdx.x * 128 + wave * 16;

  v8f acc0 = {};  // M rows 0..15
  v8f acc1 = {};  // M rows 16..31

  for (int k0 = 0; k0 < EE; k0 += 32) {
    v16bf bt = load_b_tile(Wih, EE, n0, k0, lane);
    v16bf a0 = load_a_tile(x, EE, 0,  k0, lane);
    v16bf a1 = load_a_tile(x, EE, 16, k0, lane);
    acc0 = __builtin_amdgcn_wmma_f32_16x16x32_bf16(false, a0, false, bt, (short)0, acc0, false, false);
    acc1 = __builtin_amdgcn_wmma_f32_16x16x32_bf16(false, a1, false, bt, (short)0, acc1, false, false);
  }
  for (int k0 = 0; k0 < HH; k0 += 32) {
    v16bf bt = load_b_tile(Whh, HH, n0, k0, lane);
    v16bf a0 = load_a_tile(h, HH, 0,  k0, lane);
    v16bf a1 = load_a_tile(h, HH, 16, k0, lane);
    acc0 = __builtin_amdgcn_wmma_f32_16x16x32_bf16(false, a0, false, bt, (short)0, acc0, false, false);
    acc1 = __builtin_amdgcn_wmma_f32_16x16x32_bf16(false, a1, false, bt, (short)0, acc1, false, false);
  }

  const int n    = n0 + (lane & 15);
  const float bs = b_ih[n] + b_hh[n];
  const int mb   = (lane < 16) ? 0 : 8;
#pragma unroll
  for (int r = 0; r < 8; ++r) {
    gates[(size_t)(mb + r)      * G4 + n] = acc0[r] + bs;
    gates[(size_t)(mb + r + 16) * G4 + n] = acc1[r] + bs;
  }
}

// ---- elementwise LSTM cell update ----
__global__ void __launch_bounds__(256)
k_cell(const float* __restrict__ gates, float* __restrict__ c,
       __bf16* __restrict__ h_bf) {
  int i = blockIdx.x * blockDim.x + threadIdx.x;   // 0 .. 32*1024-1
  if (i >= BB * HH) return;
  int b = i >> 10;             // HH = 1024
  int n = i & (HH - 1);
  const float* gr = gates + (size_t)b * G4;
  float gi = gr[n];
  float gf = gr[HH + n];
  float gg = gr[2 * HH + n];
  float go = gr[3 * HH + n];
  float si = 1.0f / (1.0f + expf(-gi));
  float sf = 1.0f / (1.0f + expf(-gf));
  float so = 1.0f / (1.0f + expf(-go));
  float cn = sf * c[i] + si * tanhf(gg);
  float hn = so * tanhf(cn);
  c[i]    = cn;
  h_bf[i] = (__bf16)hn;
}

// ---- fused logits GEMM + per-strip log-softmax partials --------------------
// y(32,32000) = h(32,1024) @ Wout^T + b_out, never materialized in global mem.
// Per WG: 128 vocab columns. Weights are double-buffer staged global->LDS via
// CDNA5 async-to-LDS DMA (ASYNCcnt), WMMA consumes B tiles from LDS.
// Epilogue: per-row (max, sum(exp)) over the strip + target-logit capture.
__global__ void __launch_bounds__(256)
k_logits_fused(const __bf16* __restrict__ h, const __bf16* __restrict__ Wout,
               const float* __restrict__ b_out, const int* __restrict__ idx,
               int t, float* __restrict__ pmax, float* __restrict__ psum,
               float* __restrict__ tlog) {
  // 16KB: during K loop = 2 x (128 rows x 64B) weight stage buffers;
  // after K loop      = 32 x 128 float logits scratch (aliased).
  __shared__ __align__(16) unsigned char smem[16384];
  float* red = (float*)smem;

  const int tid  = threadIdx.x;
  const int lane = tid & 31;
  const int wave = tid >> 5;
  const int n0   = blockIdx.x * 128;      // WG column strip base
  const uint32_t lds_base = (uint32_t)(uintptr_t)&smem[0];  // LDS byte address

  // Stage k-tile kt (cols n0..n0+127, k = kt*32..+31) into buffer `buf`.
  // 128 rows x 64B = 512 b128 DMAs; each of 256 lanes issues 2.
  auto issue_stage = [&](int buf, int kt) {
#pragma unroll
    for (int j = 0; j < 2; ++j) {
      const int i   = tid + j * 256;
      const int row = i >> 2;
      const int seg = i & 3;
      const __bf16* g = Wout + (size_t)(n0 + row) * HH + kt * 32 + seg * 8;
      const uint32_t l = lds_base + (uint32_t)(buf * 8192 + row * 64 + seg * 16);
      asm volatile("global_load_async_to_lds_b128 %0, %1, off"
                   :: "v"(l), "v"(g) : "memory");
    }
  };

  v8f acc0 = {};
  v8f acc1 = {};

  issue_stage(0, 0);
  for (int kt = 0; kt < HH / 32; ++kt) {
    const int cur = kt & 1;
    if (kt < HH / 32 - 1) {
      issue_stage(cur ^ 1, kt + 1);                       // overlap next DMA
      asm volatile("s_wait_asynccnt 0x2" ::: "memory");   // cur stage done
    } else {
      asm volatile("s_wait_asynccnt 0x0" ::: "memory");
    }
    __syncthreads();   // all waves' DMA into `cur` visible

    // B tile from LDS: wave's 16 rows, row-major 64B per row
    ABu b;
    {
      const int n  = lane & 15;
      const int kb = (lane < 16) ? 0 : 16;
      const __bf16* brow =
          (const __bf16*)(smem + cur * 8192 + (wave * 16 + n) * 64) + kb;
#pragma unroll
      for (int v = 0; v < 8; ++v) b.u[v] = *(const uint32_t*)(brow + 2 * v);
    }
    v16bf a0 = load_a_tile(h, HH, 0,  kt * 32, lane);
    v16bf a1 = load_a_tile(h, HH, 16, kt * 32, lane);
    acc0 = __builtin_amdgcn_wmma_f32_16x16x32_bf16(false, a0, false, b.v, (short)0, acc0, false, false);
    acc1 = __builtin_amdgcn_wmma_f32_16x16x32_bf16(false, a1, false, b.v, (short)0, acc1, false, false);
    __syncthreads();   // done reading `cur` before it is refilled (2 iters on)
  }

  // ---- epilogue: dump strip logits to LDS, reduce per row ----
  const int col = wave * 16 + (lane & 15);
  const int mb  = (lane < 16) ? 0 : 8;
  const float bs = b_out[n0 + col];
#pragma unroll
  for (int r = 0; r < 8; ++r) {
    red[(mb + r)      * 128 + col] = acc0[r] + bs;
    red[(mb + r + 16) * 128 + col] = acc1[r] + bs;
  }
  __syncthreads();

  if (tid < BB) {
    const int m   = tid;
    const int tgt = idx[(t + 1) * BB + m];
    const float* rowv = red + m * 128;
    float mx = -INFINITY;
    for (int c2 = 0; c2 < 128; ++c2) mx = fmaxf(mx, rowv[c2]);
    float s = 0.0f, tl = 0.0f;
    bool hit = false;
    for (int c2 = 0; c2 < 128; ++c2) {
      const float v = rowv[c2];
      s += expf(v - mx);
      if (n0 + c2 == tgt) { tl = v; hit = true; }
    }
    pmax[blockIdx.x * BB + m] = mx;
    psum[blockIdx.x * BB + m] = s;
    if (hit) tlog[m] = tl;     // exactly one WG hits each row's target
  }
}

// ---- combine 250 per-strip partials into the scalar loss ----
__global__ void k_loss_final(const float* __restrict__ pmax,
                             const float* __restrict__ psum,
                             const float* __restrict__ tlog,
                             const int* __restrict__ idx,
                             int t, float* __restrict__ out) {
  __shared__ float acc[BB];
  const int m = threadIdx.x;   // 32 threads, 1 block
  float mx = -INFINITY;
  for (int w = 0; w < NWG_LOGITS; ++w) mx = fmaxf(mx, pmax[w * BB + m]);
  float s = 0.0f;
  for (int w = 0; w < NWG_LOGITS; ++w)
    s += psum[w * BB + m] * expf(pmax[w * BB + m] - mx);
  const int tgt = idx[(t + 1) * BB + m];
  acc[m] = (tgt != 0) ? (mx + logf(s) - tlog[m]) : 0.0f;   // PAD == 0
  __syncthreads();
  if (m == 0) {
    float tot = 0.0f;
    for (int i = 0; i < BB; ++i) tot += acc[i];
    atomicAdd(out, tot);
  }
}

// ---- host-side orchestration ----
extern "C" void kernel_launch(void* const* d_in, const int* in_sizes, int n_in,
                              void* d_out, int out_size, void* d_ws, size_t ws_size,
                              hipStream_t stream) {
  const int*   indices = (const int*)  d_in[0];
  const float* emb     = (const float*)d_in[1];
  const float* W_ih    = (const float*)d_in[2];
  const float* W_hh    = (const float*)d_in[3];
  const float* b_ih    = (const float*)d_in[4];
  const float* b_hh    = (const float*)d_in[5];
  const float* W_out   = (const float*)d_in[6];
  const float* b_out   = (const float*)d_in[7];
  const float* h0      = (const float*)d_in[8];
  const float* c0      = (const float*)d_in[9];
  float* out = (float*)d_out;

  // workspace carve-up (256B aligned)
  char* ws = (char*)d_ws;
  size_t off = 0;
  auto take = [&](size_t bytes) { void* p = ws + off; off = (off + bytes + 255) & ~(size_t)255; return p; };
  __bf16* Wih_bf  = (__bf16*)take((size_t)G4 * EE * 2);          // 4 MB
  __bf16* Whh_bf  = (__bf16*)take((size_t)G4 * HH * 2);          // 8 MB
  __bf16* Wout_bf = (__bf16*)take((size_t)VV * HH * 2);          // 64 MB
  __bf16* xall    = (__bf16*)take((size_t)TSTEPS * BB * EE * 2); // 4 MB
  __bf16* h_bf    = (__bf16*)take((size_t)BB * HH * 2);
  float*  c_f     = (float*) take((size_t)BB * HH * 4);
  float*  gates   = (float*) take((size_t)BB * G4 * 4);
  float*  pmax    = (float*) take((size_t)NWG_LOGITS * BB * 4);  // 32 KB
  float*  psum    = (float*) take((size_t)NWG_LOGITS * BB * 4);  // 32 KB
  float*  tlog    = (float*) take((size_t)BB * 4);
  (void)ws_size; (void)in_sizes; (void)n_in; (void)out_size;

  // one-time prep
  k_zero<<<1, 1, 0, stream>>>(out);
  {
    int n = G4 * EE;
    k_cvt_bf16<<<(n + 255) / 256, 256, 0, stream>>>(W_ih, Wih_bf, n);
  }
  {
    int n = G4 * HH;
    k_cvt_bf16<<<(n + 255) / 256, 256, 0, stream>>>(W_hh, Whh_bf, n);
  }
  {
    int n = VV * HH;
    k_cvt_bf16<<<(n + 255) / 256, 256, 0, stream>>>(W_out, Wout_bf, n);
  }
  {
    int n = BB * HH;
    k_cvt_bf16<<<(n + 255) / 256, 256, 0, stream>>>(h0, h_bf, n);
  }
  hipMemcpyAsync(c_f, c0, (size_t)BB * HH * 4, hipMemcpyDeviceToDevice, stream);
  {
    int n = TSTEPS * BB * EE;
    k_embed<<<(n + 255) / 256, 256, 0, stream>>>(indices, emb, xall);
  }

  // 127 recurrent steps
  for (int t = 0; t < TSTEPS; ++t) {
    const __bf16* xt = xall + (size_t)t * BB * EE;
    k_gates<<<G4 / 128, 256, 0, stream>>>(xt, h_bf, Wih_bf, Whh_bf, b_ih, b_hh, gates);
    k_cell<<<(BB * HH) / 256, 256, 0, stream>>>(gates, c_f, h_bf);
    k_logits_fused<<<NWG_LOGITS, 256, 0, stream>>>(h_bf, Wout_bf, b_out, indices, t,
                                                   pmax, psum, tlog);
    k_loss_final<<<1, BB, 0, stream>>>(pmax, psum, tlog, indices, t, out);
  }
}
